// ResidualAGCBlock_50105088475514
// MI455X (gfx1250) — matrix operand order repaired and checked by reading, not compiled, not run on hardware
//
#include <hip/hip_runtime.h>

#define NC   32      // node feature channels
#define DEA  4       // edge attr dim
#define H1W  128
#define H2W  256
#define TE   32      // edges per block (2 WMMA M-blocks)
#define CC   1024    // C*C

typedef __attribute__((ext_vector_type(16))) _Float16 v16h;
typedef __attribute__((ext_vector_type(2)))  _Float16 v2h;
typedef __attribute__((ext_vector_type(8)))  float    v8f;

#define WMMA_F16(A, B, C) __builtin_amdgcn_wmma_f32_16x16x32_f16(false, (A), false, (B), (short)0, (C), false, false)

// ---------------------------------------------------------------------------
// A-fragment gather from row-major f16 LDS row (16-bit A 16x32 layout,
// CDNA5 ISA 7.12.2): lane m (0..15) halves 0..7 -> K=kc*32+hi*8+h,
// halves 8..15 -> K=kc*32+16+hi*8+(h-8). Contiguous halves -> ds_load_b128 x2.
// ---------------------------------------------------------------------------
__device__ __forceinline__ v16h load_afrag(const _Float16* row, int kc, int hi) {
    const _Float16* p0 = row + kc * 32 + hi * 8;
    v16h a;
#pragma unroll
    for (int j = 0; j < 4; ++j) {
        v2h lo = *(const v2h*)(p0 + 2 * j);
        v2h hh = *(const v2h*)(p0 + 16 + 2 * j);
        a[2 * j]     = lo[0]; a[2 * j + 1] = lo[1];
        a[8 + 2 * j] = hh[0]; a[9 + 2 * j] = hh[1];
    }
    return a;
}

// ---------------------------------------------------------------------------
// Pre-swizzle a row-major f32 weight [K][Nf] into f16 WMMA B-fragment order:
// frag (nt,kc) holds B tile K=kc*32..+31, N=nt*16..+15; lane l keeps 16
// halves: element h -> (k = kc*32 + (h<8 ? (l>>4)*8 + h : 16 + (l>>4)*8 + h-8), n = nt*16+(l&15)).
// ---------------------------------------------------------------------------
__global__ void prep_frag(const float* __restrict__ W, _Float16* __restrict__ out,
                          int K, int Nf) {
    int gid = blockIdx.x * 256 + threadIdx.x;
    if (gid >= K * Nf) return;
    int frag  = gid >> 9;
    int wi    = gid & 511;
    int numKC = K >> 5;
    int nt = frag / numKC, kc = frag % numKC;
    int l = wi >> 4, h = wi & 15;
    int k = kc * 32 + (h < 8 ? ((l >> 4) << 3) + h : 16 + ((l >> 4) << 3) + (h - 8));
    int n = nt * 16 + (l & 15);
    out[gid] = (_Float16)W[(size_t)k * Nf + n];
}

__global__ void zero_f32(float* p, int n) {
    int g = blockIdx.x * 256 + threadIdx.x;
    if (g < n) p[g] = 0.0f;
}

// ---------------------------------------------------------------------------
// Fused edge-conditioned conv over a 32-edge tile:
//   h1 = relu(ea@w1+b1)            (VALU, K=4)
//   h2 = relu(h1@w2+b2)            (WMMA f16, K=128, 2 M-blocks share each B)
//   theta = h2@w3+b3               (WMMA f16, K=256, 4 LDS chunks of 256 cols)
//   msg[e][o] = sum_i x[src[e]][i]*theta[e][i*32+o]   (fused per chunk)
//   atomic scatter-add msg into s[dst], count into cnt[dst]
// ---------------------------------------------------------------------------
__global__ __launch_bounds__(128) void agc_fused(
    const float* __restrict__ xin, const int* __restrict__ eidx,
    const float* __restrict__ eattr,
    const float* __restrict__ w1,  const float* __restrict__ b1,
    const _Float16* __restrict__ w2f, const float* __restrict__ b2,
    const _Float16* __restrict__ w3f, const float* __restrict__ b3,
    float* __restrict__ sacc, float* __restrict__ cnt, int E)
{
    __shared__ __align__(16) float     ea_s[TE][DEA];
    __shared__ __align__(16) float     xs_s[TE][NC];
    __shared__               int       dst_s[TE];
    __shared__ __align__(16) _Float16  h1_s[TE][H1W];
    __shared__ __align__(16) _Float16  h2_s[TE][H2W];
    __shared__ __align__(16) float     th_s[TE][256];   // quarter of theta cols

    const int tid  = threadIdx.x;
    const int lane = tid & 31, wv = tid >> 5;
    const int hi   = (lane >> 4) & 1, l15 = lane & 15;
    const int e0   = blockIdx.x * TE;

    // ---- phase 0: stage edge attrs, dst ids, gathered source features ----
    {
        int e = tid >> 2, d = tid & 3;                  // 128 threads = 32x4
        ea_s[e][d] = (e0 + e < E) ? eattr[(size_t)(e0 + e) * DEA + d] : 0.0f;
    }
    if (tid < TE) dst_s[tid] = (e0 + tid < E) ? eidx[E + e0 + tid] : -1;
#pragma unroll
    for (int it = 0; it < (TE * NC) / 128; ++it) {
        int idx = tid + it * 128;
        int e = idx >> 5, c = idx & 31;
        int s = (e0 + e < E) ? eidx[e0 + e] : 0;
        xs_s[e][c] = xin[(size_t)s * NC + c];
    }
    __syncthreads();

    // ---- phase 1: h1 = relu(ea @ w1 + b1), K=4 -> plain VALU ----
    {
        int e  = tid >> 2;                              // 0..31
        int cb = (tid & 3) * 32;                        // 0,32,64,96
        float a0 = ea_s[e][0], a1 = ea_s[e][1], a2 = ea_s[e][2], a3 = ea_s[e][3];
#pragma unroll
        for (int j = 0; j < 32; ++j) {
            int c = cb + j;
            float acc = b1[c] + a0 * w1[c] + a1 * w1[H1W + c]
                               + a2 * w1[2 * H1W + c] + a3 * w1[3 * H1W + c];
            h1_s[e][c] = (_Float16)fmaxf(acc, 0.0f);
        }
    }
    __syncthreads();

    // ---- phase 2: h2 = relu(h1 @ w2 + b2)  [32x256], WMMA K=128 ----
#pragma unroll
    for (int q = 0; q < 4; ++q) {
        int nt = wv + q * 4;                            // 16 N-tiles / 4 waves
        v8f acc0 = {}, acc1 = {};
#pragma unroll
        for (int kc = 0; kc < H1W / 32; ++kc) {
            v16h b  = *(const v16h*)(w2f + (((size_t)nt * (H1W / 32) + kc) << 9) + lane * 16);
            v16h a0 = load_afrag(&h1_s[l15][0],      kc, hi);
            v16h a1 = load_afrag(&h1_s[16 + l15][0], kc, hi);
            acc0 = WMMA_F16(a0, b, acc0);               // M-block 0 reuses b
            acc1 = WMMA_F16(a1, b, acc1);               // M-block 1
        }
        int n = nt * 16 + l15;
        float bb = b2[n];
#pragma unroll
        for (int r = 0; r < 8; ++r) {                   // D: VGPR r -> M = r + 8*hi
            h2_s[r + 8 * hi][n]      = (_Float16)fmaxf(acc0[r] + bb, 0.0f);
            h2_s[16 + r + 8 * hi][n] = (_Float16)fmaxf(acc1[r] + bb, 0.0f);
        }
    }
    __syncthreads();

    // ---- phase 3: theta = h2 @ w3 + b3 in four 256-col chunks + msg fuse ----
    float macc[8] = {0.f, 0.f, 0.f, 0.f, 0.f, 0.f, 0.f, 0.f};
    for (int ch = 0; ch < 4; ++ch) {
#pragma unroll
        for (int q = 0; q < 4; ++q) {
            int ntl = wv * 4 + q;                       // 16 tiles / 4 waves
            int nt  = ch * 16 + ntl;
            v8f acc0 = {}, acc1 = {};
#pragma unroll
            for (int kc = 0; kc < H2W / 32; ++kc) {
                v16h b  = *(const v16h*)(w3f + (((size_t)nt * (H2W / 32) + kc) << 9) + lane * 16);
                v16h a0 = load_afrag(&h2_s[l15][0],      kc, hi);
                v16h a1 = load_afrag(&h2_s[16 + l15][0], kc, hi);
                acc0 = WMMA_F16(a0, b, acc0);
                acc1 = WMMA_F16(a1, b, acc1);
            }
            int nl = ntl * 16 + l15;
            float bb = b3[nt * 16 + l15];
#pragma unroll
            for (int r = 0; r < 8; ++r) {
                th_s[r + 8 * hi][nl]      = acc0[r] + bb;
                th_s[16 + r + 8 * hi][nl] = acc1[r] + bb;
            }
        }
        __syncthreads();
        // chunk covers input channels i = ch*8 .. ch*8+7:
        // theta flat col (i*32+o) -> th_s[e][i_local*32 + o]
#pragma unroll
        for (int it = 0; it < 8; ++it) {
            int idx = tid + it * 128;                   // 1024 (e,o) pairs
            int e = idx >> 5, o = idx & 31;
            float m = macc[it];
            const float* xr = &xs_s[e][ch * 8];
            const float* tr = &th_s[e][o];
#pragma unroll
            for (int i = 0; i < 8; ++i) m += xr[i] * tr[i * 32];
            macc[it] = m;
        }
        __syncthreads();                                // th_s reused next chunk
    }

    // ---- phase 4: scatter-add to destination nodes ----
#pragma unroll
    for (int it = 0; it < 8; ++it) {
        int idx = tid + it * 128;
        int e = idx >> 5, o = idx & 31;
        if (dst_s[e] >= 0)
            atomicAdd(&sacc[(size_t)dst_s[e] * NC + o], macc[it]);
    }
    if (tid < TE && dst_s[tid] >= 0)
        atomicAdd(&cnt[dst_s[tid]], 1.0f);
}

// ---------------------------------------------------------------------------
// h[n][c] = s[n][c] / max(cnt[n], 1)
// ---------------------------------------------------------------------------
__global__ void seg_mean(const float* __restrict__ s, const float* __restrict__ cnt,
                         float* __restrict__ h, int total) {
    int g = blockIdx.x * 256 + threadIdx.x;
    if (g >= total) return;
    int n = g >> 5;
    h[g] = s[g] / fmaxf(cnt[n], 1.0f);
}

// ---------------------------------------------------------------------------
// Per-channel sum & sum-of-squares into stats[0..31], stats[32..63]
// ---------------------------------------------------------------------------
__global__ __launch_bounds__(256) void bn_stats(const float* __restrict__ h,
                                                float* __restrict__ stats, int nrows) {
    __shared__ float ls[256], lq[256];
    int tid = threadIdx.x;
    int c = tid & 31;
    float s = 0.0f, q = 0.0f;
    for (int r = blockIdx.x * 8 + (tid >> 5); r < nrows; r += gridDim.x * 8) {
        float v = h[(size_t)r * NC + c];
        s += v; q += v * v;
    }
    ls[tid] = s; lq[tid] = q; __syncthreads();
    if (tid < 128) { ls[tid] += ls[tid + 128]; lq[tid] += lq[tid + 128]; } __syncthreads();
    if (tid < 64)  { ls[tid] += ls[tid + 64];  lq[tid] += lq[tid + 64];  } __syncthreads();
    if (tid < 32) {
        atomicAdd(&stats[c],      ls[tid] + ls[tid + 32]);
        atomicAdd(&stats[32 + c], lq[tid] + lq[tid + 32]);
    }
}

__global__ void bn_apply_relu(float* __restrict__ h, const float* __restrict__ stats,
                              const float* __restrict__ g, const float* __restrict__ b,
                              int total, float invN) {
    int gid = blockIdx.x * 256 + threadIdx.x;
    if (gid >= total) return;
    int c = gid & 31;
    float m = stats[c] * invN;
    float v = stats[32 + c] * invN - m * m;
    float y = (h[gid] - m) * rsqrtf(v + 1e-5f) * g[c] + b[c];
    h[gid] = fmaxf(y, 0.0f);
}

__global__ void bn_apply_res_relu(const float* __restrict__ h, const float* __restrict__ stats,
                                  const float* __restrict__ g, const float* __restrict__ b,
                                  const float* __restrict__ x, float* __restrict__ out,
                                  int total, float invN) {
    int gid = blockIdx.x * 256 + threadIdx.x;
    if (gid >= total) return;
    int c = gid & 31;
    float m = stats[c] * invN;
    float v = stats[32 + c] * invN - m * m;
    float y = (h[gid] - m) * rsqrtf(v + 1e-5f) * g[c] + b[c];
    out[gid] = fmaxf(y + x[gid], 0.0f);
}

// ---------------------------------------------------------------------------
extern "C" void kernel_launch(void* const* d_in, const int* in_sizes, int n_in,
                              void* d_out, int out_size, void* d_ws, size_t ws_size,
                              hipStream_t stream) {
    const float* x    = (const float*)d_in[0];
    const int*   eidx = (const int*)d_in[1];
    const float* ea   = (const float*)d_in[2];
    const float* f1w1 = (const float*)d_in[3];  const float* f1b1 = (const float*)d_in[4];
    const float* f1w2 = (const float*)d_in[5];  const float* f1b2 = (const float*)d_in[6];
    const float* f1w3 = (const float*)d_in[7];  const float* f1b3 = (const float*)d_in[8];
    const float* f2w1 = (const float*)d_in[9];  const float* f2b1 = (const float*)d_in[10];
    const float* f2w2 = (const float*)d_in[11]; const float* f2b2 = (const float*)d_in[12];
    const float* f2w3 = (const float*)d_in[13]; const float* f2b3 = (const float*)d_in[14];
    const float* g1   = (const float*)d_in[15]; const float* be1  = (const float*)d_in[16];
    const float* g2   = (const float*)d_in[17]; const float* be2  = (const float*)d_in[18];
    float* out = (float*)d_out;

    const int N     = in_sizes[0] / NC;
    const int E     = in_sizes[2] / DEA;
    const int total = N * NC;
    const float invN = 1.0f / (float)N;

    // workspace carve-out (256B aligned slices)
    char* ws = (char*)d_ws;
    size_t off = 0;
    auto carve = [&](size_t bytes) -> void* {
        void* p = ws + off;
        off = (off + bytes + 255) & ~(size_t)255;
        return p;
    };
    _Float16* w2f1 = (_Float16*)carve((size_t)H1W * H2W * 2);
    _Float16* w3f1 = (_Float16*)carve((size_t)H2W * CC  * 2);
    _Float16* w2f2 = (_Float16*)carve((size_t)H1W * H2W * 2);
    _Float16* w3f2 = (_Float16*)carve((size_t)H2W * CC  * 2);
    float* sacc  = (float*)carve((size_t)total * 4);
    float* cnt   = (float*)carve((size_t)N * 4);
    float* hmid  = (float*)carve((size_t)total * 4);
    float* stats = (float*)carve(64 * 4);

    const int eblocks = (E + TE - 1) / TE;
    const int tblocks = (total + 255) / 256;

    // weight pre-swizzle to WMMA B-fragment f16 layout (lives in L2 afterwards)
    prep_frag<<<(H1W * H2W + 255) / 256, 256, 0, stream>>>(f1w2, w2f1, H1W, H2W);
    prep_frag<<<(H2W * CC  + 255) / 256, 256, 0, stream>>>(f1w3, w3f1, H2W, CC);
    prep_frag<<<(H1W * H2W + 255) / 256, 256, 0, stream>>>(f2w2, w2f2, H1W, H2W);
    prep_frag<<<(H2W * CC  + 255) / 256, 256, 0, stream>>>(f2w3, w3f2, H2W, CC);

    // ---- layer 1 ----
    zero_f32<<<tblocks, 256, 0, stream>>>(sacc, total);
    zero_f32<<<(N + 255) / 256, 256, 0, stream>>>(cnt, N);
    zero_f32<<<1, 256, 0, stream>>>(stats, 64);
    agc_fused<<<eblocks, 128, 0, stream>>>(x, eidx, ea, f1w1, f1b1, w2f1, f1b2,
                                           w3f1, f1b3, sacc, cnt, E);
    seg_mean<<<tblocks, 256, 0, stream>>>(sacc, cnt, hmid, total);
    bn_stats<<<128, 256, 0, stream>>>(hmid, stats, N);
    bn_apply_relu<<<tblocks, 256, 0, stream>>>(hmid, stats, g1, be1, total, invN);

    // ---- layer 2 ----
    zero_f32<<<tblocks, 256, 0, stream>>>(sacc, total);
    zero_f32<<<(N + 255) / 256, 256, 0, stream>>>(cnt, N);
    zero_f32<<<1, 256, 0, stream>>>(stats, 64);
    agc_fused<<<eblocks, 128, 0, stream>>>(hmid, eidx, ea, f2w1, f2b1, w2f2, f2b2,
                                           w3f2, f2b3, sacc, cnt, E);
    seg_mean<<<tblocks, 256, 0, stream>>>(sacc, cnt, hmid, total);
    bn_stats<<<128, 256, 0, stream>>>(hmid, stats, N);
    bn_apply_res_relu<<<tblocks, 256, 0, stream>>>(hmid, stats, g2, be2, x, out,
                                                   total, invN);
}
